// LlamaAttention_46059229282408
// MI455X (gfx1250) — matrix operand order repaired and checked by reading, not compiled, not run on hardware
//
#include <hip/hip_runtime.h>

// ---- CDNA5 / gfx1250: wave32, WMMA 16x16x32 bf16, TDM async tensor loads ----
typedef __attribute__((ext_vector_type(16))) __bf16       v16bf;
typedef __attribute__((ext_vector_type(8)))  float        v8f;
typedef __attribute__((ext_vector_type(4)))  unsigned int v4u;
typedef __attribute__((ext_vector_type(4)))  int          v4i;
typedef __attribute__((ext_vector_type(8)))  int          v8i;

#define S_LEN 2048
#define HID   4096
#define NH    32
#define NKV   8
#define HD    128
#define KVD   (NKV*HD)   // 1024

// ---- bf16 helpers (bitwise RNE, no FPTrunc-to-bf16 dependence) ----
__device__ inline unsigned short bfb(float f) {
  unsigned u = __float_as_uint(f);
  unsigned r = u + 0x7FFFu + ((u >> 16) & 1u);
  return (unsigned short)(r >> 16);
}
__device__ inline __bf16 f2bf(float f) {
  unsigned short h = bfb(f);
  __bf16 out;
  __builtin_memcpy(&out, &h, sizeof(out));
  return out;
}
__device__ inline float bf2f(__bf16 b) {
  unsigned short h;
  __builtin_memcpy(&h, &b, sizeof(h));
  return __uint_as_float(((unsigned)h) << 16);
}
__device__ inline unsigned pk2(float lo, float hi) {
  return (unsigned)bfb(lo) | ((unsigned)bfb(hi) << 16);
}

// A-matrix (16-bit, 16x32) per-lane element j -> K index (ISA 7.12.2)
__device__ inline int akmap(int j, int half) { return j + 8 * half + ((j >= 8) ? 8 : 0); }

__device__ inline v8f bwmma(v16bf a, v16bf b, v8f c) {
  return __builtin_amdgcn_wmma_f32_16x16x32_bf16(false, a, false, b, (short)0, c, false, false);
}

// ---- Tensor Data Mover: DMA a 32x128 bf16 tile ([rows][1024-elem stride]) into LDS,
//      TDM applies 16B padding per 256B row -> LDS row stride 136 halfs (ISA ch.8 D#).
__device__ __forceinline__ void tdm_load_tile32x128(const __bf16* gbase, unsigned ldsAddr) {
  unsigned long long ga = (unsigned long long)(const void*)gbase;
  v4u g0;
  g0[0] = 1u;                                   // count=1, user descriptor, no gather
  g0[1] = ldsAddr;                              // lds_addr (bytes)
  g0[2] = (unsigned)ga;                         // global_addr[31:0]
  g0[3] = ((unsigned)(ga >> 32) & 0x01FFFFFFu)  // global_addr[56:32]
        | 0x80000000u;                          // type=2 ("image")
  v8i g1;
  g1[0] = (int)((1u << 16)     // data_size = 2 bytes
              | (1u << 20)     // pad_enable
              | (5u << 22)     // pad_interval: every 64 dwords (=256B row)
              | (3u << 25));   // pad_amount: 4 dwords (=16B) -> 136-half stride
  g1[1] = (int)(((unsigned)KVD & 0xFFFFu) << 16);                      // tensor_dim0 lo
  g1[2] = (int)((((unsigned)KVD >> 16) & 0xFFFFu)
              | (((unsigned)S_LEN & 0xFFFFu) << 16));                  // dim0 hi | dim1 lo
  g1[3] = (int)((((unsigned)S_LEN >> 16) & 0xFFFFu) | (128u << 16));   // dim1 hi | tile_dim0=128
  g1[4] = (int)(32u);                                                  // tile_dim1=32, tile_dim2=0
  g1[5] = (int)((unsigned)KVD);                                        // tensor_dim0_stride lo
  g1[6] = 0;                                                           // stride0 hi | stride1 lo
  g1[7] = 0;
  v4i z4 = {};
#if defined(__clang_major__) && __clang_major__ >= 23
  v8i z8 = {};
  __builtin_amdgcn_tensor_load_to_lds(g0, g1, z4, z4, z8, 0);
#else
  __builtin_amdgcn_tensor_load_to_lds(g0, g1, z4, z4, 0);
#endif
}

// ======================= GEMM: C[M,N] = A[M,K] * W[N,K]^T =======================
// 128x128 block tile, 8 waves, each wave 32x64 (8 WMMA / K-step),
// float4 global loads + LDS double buffering.
#define BK  32
#define LDT 36   // padded LDS row stride (bf16 elems); rows 8B-aligned for ds_store_b64

template <bool OUT_BF16>
__global__ __launch_bounds__(256) void gemm_xwt(
    const float* __restrict__ A, const float* __restrict__ W,
    void* __restrict__ Cv, int M, int N, int K)
{
  __shared__ __bf16 As[2 * 128 * LDT];
  __shared__ __bf16 Ws[2 * 128 * LDT];

  const int tid   = threadIdx.x;
  const int lane  = tid & 31;
  const int wid   = tid >> 5;
  const int waveM = wid & 3;     // 4 m-groups of 32 rows
  const int waveN = wid >> 2;    // 2 n-groups of 64 cols
  const int mBase = blockIdx.y * 128;
  const int nBase = blockIdx.x * 128;
  const int half  = lane >> 4;
  const int l16   = lane & 15;

  // staging map: 128x32 tile = 1024 float4 loads, 4 per thread
  int srow[4], sc4[4];
  #pragma unroll
  for (int i = 0; i < 4; ++i) {
    int idx = tid + i * 256;
    srow[i] = idx >> 3;        // 8 float4 per row
    sc4[i]  = idx & 7;
  }

  float4 ar[4], wr[4];
  auto stageLoad = [&](int kt) {
    #pragma unroll
    for (int i = 0; i < 4; ++i) {
      ar[i] = *(const float4*)(A + (size_t)(mBase + srow[i]) * K + (size_t)kt * BK + sc4[i] * 4);
      wr[i] = *(const float4*)(W + (size_t)(nBase + srow[i]) * K + (size_t)kt * BK + sc4[i] * 4);
    }
  };
  auto stageStore = [&](int buf) {
    __bf16* ab = &As[buf * 128 * LDT];
    __bf16* wb = &Ws[buf * 128 * LDT];
    #pragma unroll
    for (int i = 0; i < 4; ++i) {
      uint2 pa; pa.x = pk2(ar[i].x, ar[i].y); pa.y = pk2(ar[i].z, ar[i].w);
      uint2 pw; pw.x = pk2(wr[i].x, wr[i].y); pw.y = pk2(wr[i].z, wr[i].w);
      *(uint2*)&ab[srow[i] * LDT + sc4[i] * 4] = pa;
      *(uint2*)&wb[srow[i] * LDT + sc4[i] * 4] = pw;
    }
  };

  v8f acc[2][4] = {};

  stageLoad(0);
  stageStore(0);
  __syncthreads();

  const int nk = K / BK;
  for (int kt = 0; kt < nk; ++kt) {
    if (kt + 1 < nk) stageLoad(kt + 1);   // global loads overlap WMMAs below

    const int buf = kt & 1;
    const __bf16* ab = &As[buf * 128 * LDT];
    const __bf16* wb = &Ws[buf * 128 * LDT];

    v16bf a0, a1;
    #pragma unroll
    for (int j = 0; j < 16; ++j) {
      int k = akmap(j, half);
      a0[j] = ab[(waveM * 32 +      l16) * LDT + k];
      a1[j] = ab[(waveM * 32 + 16 + l16) * LDT + k];
    }
    #pragma unroll
    for (int c = 0; c < 4; ++c) {
      v16bf b;
      #pragma unroll
      for (int j = 0; j < 16; ++j)
        b[j] = wb[(waveN * 64 + c * 16 + l16) * LDT + akmap(j, half)];
      acc[0][c] = bwmma(a0, b, acc[0][c]);
      acc[1][c] = bwmma(a1, b, acc[1][c]);
    }

    if (kt + 1 < nk) stageStore((kt + 1) & 1);
    __syncthreads();
  }

  // C layout: VGPR r -> row r + 8*half, col = lane&15
  #pragma unroll
  for (int mi = 0; mi < 2; ++mi) {
    const int mRow = mBase + waveM * 32 + mi * 16 + 8 * half;
    #pragma unroll
    for (int r = 0; r < 8; ++r) {
      #pragma unroll
      for (int c = 0; c < 4; ++c) {
        size_t off = (size_t)(mRow + r) * N + nBase + waveN * 64 + c * 16 + l16;
        float v = acc[mi][c][r];
        if (OUT_BF16) ((__bf16*)Cv)[off] = f2bf(v);
        else          ((float*)Cv)[off]  = v;
      }
    }
  }
}

// ======================= RoPE (in place, bf16) =======================
__global__ void rope_kernel(__bf16* __restrict__ x, const int* __restrict__ pos,
                            int nheads, int total)
{
  int idx = blockIdx.x * blockDim.x + threadIdx.x;
  if (idx >= total) return;
  int d = idx & 63;
  int t = idx >> 6;
  int h = t % nheads;
  int s = t / nheads;
  float p   = (float)pos[s];
  float inv = __expf(-(2.0f * (float)d / 128.0f) * __logf(10000.0f));
  float c, sn;
  __sincosf(p * inv, &c, &sn);
  __bf16* base = x + ((size_t)s * nheads + h) * HD;
  float x1 = bf2f(base[d]), x2 = bf2f(base[d + 64]);
  base[d]      = f2bf(x1 * c - x2 * sn);
  base[d + 64] = f2bf(x2 * c + x1 * sn);
}

// ======================= Flash attention (causal, GQA 4:1) =======================
// K/V tiles DMA'd into LDS by the Tensor Data Mover, double-buffered so the DMA of
// block it+1 overlaps the WMMAs of block it. Q/K/V are bf16 in HBM.
#define KLD (HD + 8)   // 136-half padded stride; produced by the TDM pad fields

__global__ __launch_bounds__(128) void flash_attn_kernel(
    const __bf16* __restrict__ Q, const __bf16* __restrict__ Kg,
    const __bf16* __restrict__ Vg, float* __restrict__ Out)
{
  __shared__ __bf16 Ks[2 * 32 * KLD];
  __shared__ __bf16 Vs[2 * 32 * KLD];
  __shared__ __bf16 Ps[4 * 16 * 34];  // per-wave P bounce (C-layout -> A-layout)

  const int head  = blockIdx.x;
  const int qb    = blockIdx.y;
  const int tid   = threadIdx.x;
  const int lane  = tid & 31;
  const int wave  = tid >> 5;
  const int half  = lane >> 4;
  const int l16   = lane & 15;
  const int kvH   = head >> 2;            // NH/NKV = 4
  const int qRow0 = qb * 64 + wave * 16;  // this wave's 16 q rows
  const float scale = 0.08838834764831845f; // 1/sqrt(128)

  // Q fragments: 16x128 = 4 slices of 16x32 (A layout), direct bf16 loads
  v16bf qf[4];
  #pragma unroll
  for (int sl = 0; sl < 4; ++sl)
    #pragma unroll
    for (int j = 0; j < 16; ++j)
      qf[sl][j] = Q[(size_t)(qRow0 + l16) * HID + head * HD + sl * 32 + akmap(j, half)];

  v8f o[8] = {};
  float mx[8], lsum[8];
  #pragma unroll
  for (int i = 0; i < 8; ++i) { mx[i] = -1e30f; lsum[i] = 0.0f; }

  __bf16* Pw = &Ps[wave * 16 * 34];
  const int nIter = 2 * qb + 2;   // 32 keys per iteration up to the block's causal limit

  // prologue: DMA first K/V tiles into buffer 0 (wave 0 issues; TDM ignores EXEC)
  if (wave == 0) {
    tdm_load_tile32x128(Kg + (size_t)kvH * HD, (unsigned)(size_t)&Ks[0]);
    tdm_load_tile32x128(Vg + (size_t)kvH * HD, (unsigned)(size_t)&Vs[0]);
  }

  for (int it = 0; it < nIter; ++it) {
    const int cur = it & 1;
    if (wave == 0) {
      if (it + 1 < nIter) {  // issue next tiles, then wait only for the current pair
        const size_t goff = (size_t)(it + 1) * 32 * KVD + (size_t)kvH * HD;
        tdm_load_tile32x128(Kg + goff, (unsigned)(size_t)&Ks[(cur ^ 1) * 32 * KLD]);
        tdm_load_tile32x128(Vg + goff, (unsigned)(size_t)&Vs[(cur ^ 1) * 32 * KLD]);
        __builtin_amdgcn_s_wait_tensorcnt(2);
      } else {
        __builtin_amdgcn_s_wait_tensorcnt(0);
      }
    }
    __syncthreads();

    const __bf16* Kc = &Ks[cur * 32 * KLD];
    const __bf16* Vc = &Vs[cur * 32 * KLD];
    const int kb = it * 32;

    // scores: Q(16x128) x K^T(128x32) -> two 16x16 tiles, 4 chained WMMAs each
    v8f s0 = {}, s1 = {};
    #pragma unroll
    for (int sl = 0; sl < 4; ++sl) {
      v16bf b0, b1;
      #pragma unroll
      for (int j = 0; j < 16; ++j) {
        b0[j] = Kc[(     l16) * KLD + sl * 32 + akmap(j, half)];
        b1[j] = Kc[(16 + l16) * KLD + sl * 32 + akmap(j, half)];
      }
      s0 = bwmma(qf[sl], b0, s0);
      s1 = bwmma(qf[sl], b1, s1);
    }

    // fp32 online softmax over the 32 keys (row groups = 16-lane halves)
    const int qrowBase = qRow0 + 8 * half;
    const int key0 = kb + l16, key1 = kb + 16 + l16;
    #pragma unroll
    for (int r = 0; r < 8; ++r) {
      int qr = qrowBase + r;
      float a0 = (key0 <= qr) ? s0[r] * scale : -1e30f;
      float a1 = (key1 <= qr) ? s1[r] * scale : -1e30f;
      float ml = fmaxf(a0, a1);
      #pragma unroll
      for (int off = 8; off >= 1; off >>= 1)
        ml = fmaxf(ml, __shfl_xor(ml, off, 32));
      float mnew = fmaxf(mx[r], ml);
      float p0 = __expf(a0 - mnew);
      float p1 = __expf(a1 - mnew);
      float ps = p0 + p1;
      #pragma unroll
      for (int off = 8; off >= 1; off >>= 1)
        ps += __shfl_xor(ps, off, 32);
      float corr = __expf(mx[r] - mnew);
      lsum[r] = lsum[r] * corr + ps;
      mx[r] = mnew;
      #pragma unroll
      for (int fs = 0; fs < 8; ++fs) o[fs][r] = o[fs][r] * corr;
      Pw[(r + 8 * half) * 34 +      l16] = f2bf(p0);
      Pw[(r + 8 * half) * 34 + 16 + l16] = f2bf(p1);
    }

    // P(16x32) x V(32x128): reload P in A layout, 8 WMMAs over feature slices
    v16bf pa;
    #pragma unroll
    for (int j = 0; j < 16; ++j)
      pa[j] = Pw[l16 * 34 + akmap(j, half)];
    #pragma unroll
    for (int fs = 0; fs < 8; ++fs) {
      v16bf vb;
      #pragma unroll
      for (int j = 0; j < 16; ++j)
        vb[j] = Vc[akmap(j, half) * KLD + fs * 16 + l16];
      o[fs] = bwmma(pa, vb, o[fs]);
    }
    __syncthreads();   // all reads of buf `cur` done before TDM overwrites it next iter
  }

  // normalize and store fp32 [S, NH*HD]
  #pragma unroll
  for (int r = 0; r < 8; ++r) {
    float inv = (lsum[r] > 0.0f) ? 1.0f / lsum[r] : 0.0f;
    int row = qRow0 + 8 * half + r;
    #pragma unroll
    for (int fs = 0; fs < 8; ++fs)
      Out[(size_t)row * HID + head * HD + fs * 16 + l16] = o[fs][r] * inv;
  }
}

// ======================= launch =======================
extern "C" void kernel_launch(void* const* d_in, const int* in_sizes, int n_in,
                              void* d_out, int out_size, void* d_ws, size_t ws_size,
                              hipStream_t stream) {
  (void)in_sizes; (void)n_in; (void)out_size; (void)ws_size;
  const float* X   = (const float*)d_in[0];
  const int*   pos = (const int*)d_in[1];
  const float* Wq  = (const float*)d_in[2];
  const float* Wk  = (const float*)d_in[3];
  const float* Wv  = (const float*)d_in[4];
  const float* Wo  = (const float*)d_in[5];
  float* out = (float*)d_out;

  char* ws = (char*)d_ws;
  __bf16* Qb = (__bf16*)ws;                                  // [2048,4096] bf16
  __bf16* Kb = (__bf16*)(ws + (size_t)S_LEN * HID * 2);      // [2048,1024] bf16
  __bf16* Vb = Kb + (size_t)S_LEN * KVD;                     // [2048,1024] bf16
  float*  Ab = (float*)(ws + (size_t)S_LEN * HID * 2 + (size_t)S_LEN * KVD * 4);  // [2048,4096] f32

  // QKV projections (bf16 WMMA GEMMs, bf16 outputs for the TDM attention path)
  gemm_xwt<true><<<dim3(HID / 128, S_LEN / 128), dim3(256), 0, stream>>>(X, Wq, Qb, S_LEN, HID, HID);
  gemm_xwt<true><<<dim3(KVD / 128, S_LEN / 128), dim3(256), 0, stream>>>(X, Wk, Kb, S_LEN, KVD, HID);
  gemm_xwt<true><<<dim3(KVD / 128, S_LEN / 128), dim3(256), 0, stream>>>(X, Wv, Vb, S_LEN, KVD, HID);

  // RoPE on Q and K (bf16 in place)
  {
    int totalQ = S_LEN * NH * 64;
    rope_kernel<<<(totalQ + 255) / 256, 256, 0, stream>>>(Qb, pos, NH, totalQ);
    int totalK = S_LEN * NKV * 64;
    rope_kernel<<<(totalK + 255) / 256, 256, 0, stream>>>(Kb, pos, NKV, totalK);
  }

  // causal flash attention with GQA (32 heads x 32 q-blocks), TDM-staged K/V
  flash_attn_kernel<<<dim3(NH, S_LEN / 64), dim3(128), 0, stream>>>(Qb, Kb, Vb, Ab);

  // output projection (fp32 out)
  gemm_xwt<false><<<dim3(HID / 128, S_LEN / 128), dim3(256), 0, stream>>>(Ab, Wo, out, S_LEN, HID, HID);
}
